// RealVectorSHT_26448408609073
// MI455X (gfx1250) — matrix operand order repaired and checked by reading, not compiled, not run on hardware
//
#include <hip/hip_runtime.h>
#include <stdint.h>

// RealVectorSHT on gfx1250 (MI455X), bf16 split-precision WMMA pipeline:
//  trig_init : build bf16 hi/lo DFT matrices (L2-resident, ~2.2MB)
//  split_x   : x fp32 -> zero-padded bf16 hi/lo planes (split ONCE)
//  dft_gemm  : DFT-as-GEMM; async-to-LDS bf16 tiles, 0 conversion VALU in loop;
//              C scattered into packed (hi|lo) uint32 vws [m][comp][b][k]
//  leg_gemm  : per-m Legendre GEMMs; W split once per WG into LDS; sign recombine
// Workspace (~280 MB):
//   Thi,Tlo : 768*736 ushort each
//   xhi,xlo : 46080*736 ushort each (n padded to 736, zero-filled)
//   vws     : 361*256*384 uint32 (bf16 hi | bf16 lo << 16)

typedef __attribute__((ext_vector_type(16))) __bf16 v16bf;
typedef __attribute__((ext_vector_type(8)))  float  v8f;

#define NLAT   360
#define NLON   720
#define LMAX   359
#define MMAX   361
#define NBATCH 64
#define RROWS  46080         // 64*2*360 FFT rows
#define MC     722           // 2*MMAX trig rows (re, im interleaved)
#define TROWS  768           // padded trig rows
#define TCOLS  736           // padded NLON (23 * 32)
#define KP     384           // padded NLAT (12 * 32)

__device__ __forceinline__ unsigned f2bf(float f) {
  unsigned u = __float_as_uint(f);
  return (u + 0x7FFFu + ((u >> 16) & 1u)) >> 16;   // RNE to bf16 bits
}
__device__ __forceinline__ float bf2f(unsigned b) { return __uint_as_float(b << 16); }

// split-precision accumulate: acc += (Ah+Al)*(Bh+Bl) minus the Al*Bl term
__device__ __forceinline__ v8f wmma3(v8f acc, v16bf ah, v16bf al, v16bf bh, v16bf bl) {
  acc = __builtin_amdgcn_wmma_f32_16x16x32_bf16(false, ah, false, bh, (short)0, acc, false, false);
  acc = __builtin_amdgcn_wmma_f32_16x16x32_bf16(false, ah, false, bl, (short)0, acc, false, false);
  acc = __builtin_amdgcn_wmma_f32_16x16x32_bf16(false, al, false, bh, (short)0, acc, false, false);
  return acc;
}

// ---------------------------------------------------------------- trig init
__global__ void __launch_bounds__(256) trig_init(unsigned short* Thi, unsigned short* Tlo) {
  int i = blockIdx.x * 256 + threadIdx.x;        // over TROWS*TCOLS
  int row = i / TCOLS, col = i % TCOLS;
  float v = 0.f;
  if (row < MC && col < NLON) {
    int mm = row >> 1, ri = row & 1;
    long long tmod = ((long long)mm * (long long)col) % NLON;  // exact periodic reduction
    const double TWO_PI = 6.283185307179586476925286766559;
    double ang = TWO_PI * (double)tmod / (double)NLON;
    double sc  = TWO_PI / (double)NLON;
    v = (float)(sc * (ri ? -sin(ang) : cos(ang)));
  }
  unsigned h = f2bf(v);
  unsigned e = f2bf(v - bf2f(h));
  Thi[i] = (unsigned short)h;
  Tlo[i] = (unsigned short)e;
}

// ---------------------------------------------------------------- x pre-split
// x (fp32, rows of 720) -> xhi/xlo bf16 planes with rows padded to 736 (zeros)
__global__ void __launch_bounds__(256) split_x(const float* __restrict__ x,
                                               unsigned short* __restrict__ xhi,
                                               unsigned short* __restrict__ xlo) {
  size_t idx = (size_t)blockIdx.x * 256 + threadIdx.x;   // over RROWS * (TCOLS/2)
  size_t r = idx / (TCOLS / 2);
  int col = (int)(idx % (TCOLS / 2)) * 2;
  float2 v = make_float2(0.f, 0.f);
  if (col < NLON) v = *(const float2*)&x[r * NLON + col];
  unsigned h0 = f2bf(v.x), h1 = f2bf(v.y);
  unsigned e0 = f2bf(v.x - bf2f(h0)), e1 = f2bf(v.y - bf2f(h1));
  ((unsigned*)xhi)[idx] = h0 | (h1 << 16);
  ((unsigned*)xlo)[idx] = e0 | (e1 << 16);
}

// ---------------------------------------------------------------- stage A
// C[trigRow][r] = sum_n T[trigRow][n] * x[r][n];  r = ((b*2+c)*360 + k)
// grid (12 M-chunks FAST, 360 N-chunks SLOW) so x tiles are shared in L2.
// 256 thr = 8 waves (2M x 4N); inner loop has NO conversion VALU.
__global__ void __launch_bounds__(256) dft_gemm(const unsigned short* __restrict__ xhi,
                                                const unsigned short* __restrict__ xlo,
                                                const unsigned short* __restrict__ Thi,
                                                const unsigned short* __restrict__ Tlo,
                                                unsigned* __restrict__ vws) {
  __shared__ unsigned short Xlds[2 * 128 * 40];  // [plane][row][40 bf16], 20.5KB
  const int t = threadIdx.x;
  const int lane = t & 31, wave = t >> 5;
  const int waveM = wave >> 2, waveN = wave & 3;
  const int wgM = blockIdx.x, wgN = blockIdx.y;
  const int r0 = wgN * 128;
  const int lcol = lane & 15;
  const int hf   = lane >> 4;
  const int aklo = hf * 16;                      // A-frag: 16 consecutive K
  const int bkb  = hf * 16;                      // B-frag: 16 consecutive K

  v8f acc[2][2];
#pragma unroll
  for (int i = 0; i < 2; ++i)
#pragma unroll
    for (int j = 0; j < 2; ++j) acc[i][j] = v8f{0.f,0.f,0.f,0.f,0.f,0.f,0.f,0.f};

  for (int kc = 0; kc < TCOLS; kc += 32) {
    __syncthreads();
    // async copy of bf16 hi+lo x tiles -> LDS (ASYNCcnt path); pad pre-zeroed
#pragma unroll
    for (int it = 0; it < 4; ++it) {
      int plane = it >> 1;
      int rowseg = (it & 1) * 256 + t;           // 512 16B-segments per plane
      int row = rowseg >> 2, seg = rowseg & 3;
      unsigned vOff = (unsigned)((r0 + row) * (TCOLS * 2) + kc * 2 + seg * 16);
      unsigned ldsA = (unsigned)(uintptr_t)&Xlds[(plane * 128 + row) * 40 + seg * 8];
      const unsigned short* src = plane ? xlo : xhi;
      asm volatile("global_load_async_to_lds_b128 %0, %1, %2"
                   :: "v"(ldsA), "v"(vOff), "s"(src) : "memory");
    }
    asm volatile("s_wait_asynccnt 0x0" ::: "memory");
    __syncthreads();

    // A fragments: trig bf16 planes direct from global (L2 resident)
    v16bf Ah[2], Al[2];
#pragma unroll
    for (int mt = 0; mt < 2; ++mt) {
      int row = wgM * 64 + waveM * 32 + mt * 16 + lcol;
      size_t o = (size_t)row * TCOLS + kc + aklo;
      union { v16bf v; uint4 q[2]; } Fh, Fl;
      Fh.q[0] = *(const uint4*)(Thi + o);
      Fh.q[1] = *(const uint4*)(Thi + o + 8);
      Fl.q[0] = *(const uint4*)(Tlo + o);
      Fl.q[1] = *(const uint4*)(Tlo + o + 8);
      Ah[mt] = Fh.v; Al[mt] = Fl.v;
    }

    // B fragments: straight b128 LDS reads (pre-split bf16), then 12 WMMAs
#pragma unroll
    for (int nt = 0; nt < 2; ++nt) {
      int rl = waveN * 32 + nt * 16 + lcol;
      union { v16bf v; uint4 q[2]; } Bh, Bl;
      const unsigned short* ph = &Xlds[(0 * 128 + rl) * 40 + bkb];
      const unsigned short* pl = &Xlds[(1 * 128 + rl) * 40 + bkb];
      Bh.q[0] = *(const uint4*)ph; Bh.q[1] = *(const uint4*)(ph + 8);
      Bl.q[0] = *(const uint4*)pl; Bl.q[1] = *(const uint4*)(pl + 8);
#pragma unroll
      for (int mt = 0; mt < 2; ++mt)
        acc[mt][nt] = wmma3(acc[mt][nt], Ah[mt], Al[mt], Bh.v, Bl.v);
    }
  }

  // scatter C into packed vws [m][comp][b][k]: split ONCE here at the store
#pragma unroll
  for (int mt = 0; mt < 2; ++mt) {
#pragma unroll
    for (int nt = 0; nt < 2; ++nt) {
      int r = r0 + waveN * 32 + nt * 16 + lcol;
      int b = r / 720, cch = (r / 360) & 1, k = r % 360;
#pragma unroll
      for (int j = 0; j < 8; ++j) {
        int gRow = wgM * 64 + waveM * 32 + mt * 16 + j + hf * 8;
        if (gRow < MC) {
          int mm = gRow >> 1, ri = gRow & 1;
          float el = acc[mt][nt][j];
          unsigned h = f2bf(el);
          unsigned e = f2bf(el - bf2f(h));
          vws[(((size_t)mm * 4 + (cch * 2 + ri)) * NBATCH + b) * KP + k] = h | (e << 16);
        }
      }
    }
  }
}

// ---------------------------------------------------------------- stage B
// per m: P0 = V*W0^T, P1 = V*W1^T (V rows = {ar,ai,br,bi} x 64 batch, K = nlat)
// grid (6 l-chunks FAST, 361 m SLOW) so each m's vws slice stays in L2.
__global__ void __launch_bounds__(256) leg_gemm(const unsigned* __restrict__ vws,
                                                const float* __restrict__ wgt,
                                                float* __restrict__ outp) {
  __shared__ unsigned short Wlds[4 * 64 * 40];   // [s*2+plane][l][40 bf16], 20.5KB
  __shared__ float Ps[2 * 256 * 33];             // P0 | P1, 67.6KB
  const int t = threadIdx.x, lane = t & 31, wave = t >> 5;
  const int m = blockIdx.y, lbase = blockIdx.x * 64;
  const int lcol = lane & 15, hf = lane >> 4;
  const int aklo = hf * 16;
  const int bkb  = hf * 16;

  v8f p0[2][4], p1[2][4];
#pragma unroll
  for (int i = 0; i < 2; ++i)
#pragma unroll
    for (int j = 0; j < 4; ++j) {
      p0[i][j] = v8f{0.f,0.f,0.f,0.f,0.f,0.f,0.f,0.f};
      p1[i][j] = v8f{0.f,0.f,0.f,0.f,0.f,0.f,0.f,0.f};
    }

  for (int kc = 0; kc < KP; kc += 32) {
    __syncthreads();
    // cooperative W tile: split each weight element ONCE per WG into LDS
#pragma unroll
    for (int cc = 0; cc < 4; ++cc) {
      int c = cc * 256 + t;                      // [s][l(64)][k4 chunk(8)]
      int s = c >> 9, l = (c >> 3) & 63, k4 = (c & 7) * 4;
      float4 v = make_float4(0.f, 0.f, 0.f, 0.f);
      int lg = lbase + l, kg = kc + k4;
      if (lg < LMAX && kg < NLAT)
        v = *(const float4*)&wgt[(((size_t)s * MMAX + m) * LMAX + lg) * NLAT + kg];
      unsigned h0 = f2bf(v.x), h1 = f2bf(v.y), h2 = f2bf(v.z), h3 = f2bf(v.w);
      unsigned e0 = f2bf(v.x - bf2f(h0)), e1 = f2bf(v.y - bf2f(h1));
      unsigned e2 = f2bf(v.z - bf2f(h2)), e3 = f2bf(v.w - bf2f(h3));
      unsigned* hrow = (unsigned*)&Wlds[((s * 2 + 0) * 64 + l) * 40];
      unsigned* lrow = (unsigned*)&Wlds[((s * 2 + 1) * 64 + l) * 40];
      hrow[k4 / 2]     = h0 | (h1 << 16);
      hrow[k4 / 2 + 1] = h2 | (h3 << 16);
      lrow[k4 / 2]     = e0 | (e1 << 16);
      lrow[k4 / 2 + 1] = e2 | (e3 << 16);
    }
    __syncthreads();

    // A fragments from packed vws: cheap pair-unpack (perm-able), no re-split
    v16bf Ah[2], Al[2];
#pragma unroll
    for (int mt = 0; mt < 2; ++mt) {
      int row = wave * 32 + mt * 16 + lcol;
      const unsigned* vb = vws + ((size_t)m * 256 + row) * KP;
      unsigned pk[16];
      int k0 = kc + aklo, k1 = k0 + 8;
      if (k0 < NLAT) {
        uint4 a = *(const uint4*)(vb + k0), b2 = *(const uint4*)(vb + k0 + 4);
        pk[0]=a.x; pk[1]=a.y; pk[2]=a.z; pk[3]=a.w; pk[4]=b2.x; pk[5]=b2.y; pk[6]=b2.z; pk[7]=b2.w;
      } else {
#pragma unroll
        for (int j = 0; j < 8; ++j) pk[j] = 0u;
      }
      if (k1 < NLAT) {
        uint4 a = *(const uint4*)(vb + k1), b2 = *(const uint4*)(vb + k1 + 4);
        pk[8]=a.x; pk[9]=a.y; pk[10]=a.z; pk[11]=a.w; pk[12]=b2.x; pk[13]=b2.y; pk[14]=b2.z; pk[15]=b2.w;
      } else {
#pragma unroll
        for (int j = 8; j < 16; ++j) pk[j] = 0u;
      }
      union { v16bf v; unsigned u[8]; } H, L;
#pragma unroll
      for (int j = 0; j < 8; ++j) {
        unsigned a = pk[2 * j], b2 = pk[2 * j + 1];
        H.u[j] = (a & 0xFFFFu) | (b2 << 16);
        L.u[j] = (a >> 16) | (b2 & 0xFFFF0000u);
      }
      Ah[mt] = H.v; Al[mt] = L.v;
    }

#pragma unroll
    for (int nt = 0; nt < 4; ++nt) {
      int lloc = nt * 16 + lcol;
      union { v16bf v; uint4 q[2]; } W0h, W0l, W1h, W1l;
      const unsigned short* p;
      p = &Wlds[(0 * 64 + lloc) * 40 + bkb]; W0h.q[0] = *(const uint4*)p; W0h.q[1] = *(const uint4*)(p + 8);
      p = &Wlds[(1 * 64 + lloc) * 40 + bkb]; W0l.q[0] = *(const uint4*)p; W0l.q[1] = *(const uint4*)(p + 8);
      p = &Wlds[(2 * 64 + lloc) * 40 + bkb]; W1h.q[0] = *(const uint4*)p; W1h.q[1] = *(const uint4*)(p + 8);
      p = &Wlds[(3 * 64 + lloc) * 40 + bkb]; W1l.q[0] = *(const uint4*)p; W1l.q[1] = *(const uint4*)(p + 8);
#pragma unroll
      for (int mt = 0; mt < 2; ++mt) {
        p0[mt][nt] = wmma3(p0[mt][nt], Ah[mt], Al[mt], W0h.v, W0l.v);
        p1[mt][nt] = wmma3(p1[mt][nt], Ah[mt], Al[mt], W1h.v, W1l.v);
      }
    }
  }

  // recombine o0/o1 through LDS (two 32-col halves), write interleaved complex out
  float* PS0 = Ps;
  float* PS1 = Ps + 256 * 33;
#pragma unroll
  for (int h = 0; h < 2; ++h) {
    __syncthreads();
#pragma unroll
    for (int mt = 0; mt < 2; ++mt) {
#pragma unroll
      for (int ntl = 0; ntl < 2; ++ntl) {
        int nt = 2 * h + ntl;
#pragma unroll
        for (int j = 0; j < 8; ++j) {
          int row = wave * 32 + mt * 16 + j + hf * 8;
          int col = ntl * 16 + lcol;
          PS0[row * 33 + col] = p0[mt][nt][j];
          PS1[row * 33 + col] = p1[mt][nt][j];
        }
      }
    }
    __syncthreads();
#pragma unroll
    for (int it = 0; it < 8; ++it) {
      int idx = it * 256 + t;
      int lc = idx & 31, b = idx >> 5;
      float p0ar = PS0[b * 33 + lc],         p0ai = PS0[(64 + b) * 33 + lc];
      float p0br = PS0[(128 + b) * 33 + lc], p0bi = PS0[(192 + b) * 33 + lc];
      float p1ar = PS1[b * 33 + lc],         p1ai = PS1[(64 + b) * 33 + lc];
      float p1br = PS1[(128 + b) * 33 + lc], p1bi = PS1[(192 + b) * 33 + lc];
      float o0r =  p0ar - p1bi, o0i =  p0ai + p1br;
      float o1r = -p1ai - p0br, o1i =  p1ar - p0bi;
      int l = lbase + h * 32 + lc;
      if (l < LMAX) {
        size_t i0 = (((size_t)b * 2 + 0) * LMAX + l) * MMAX + m;
        size_t i1 = (((size_t)b * 2 + 1) * LMAX + l) * MMAX + m;
        *(float2*)&outp[i0 * 2] = make_float2(o0r, o0i);
        *(float2*)&outp[i1 * 2] = make_float2(o1r, o1i);
      }
    }
  }
}

// ---------------------------------------------------------------- launcher
extern "C" void kernel_launch(void* const* d_in, const int* in_sizes, int n_in,
                              void* d_out, int out_size, void* d_ws, size_t ws_size,
                              hipStream_t stream) {
  (void)in_sizes; (void)n_in; (void)out_size; (void)ws_size;
  const float* x   = (const float*)d_in[0];
  const float* wgt = (const float*)d_in[1];
  float* outp = (float*)d_out;

  unsigned short* Thi = (unsigned short*)d_ws;
  unsigned short* Tlo = Thi + (size_t)TROWS * TCOLS;
  unsigned short* xhi = Tlo + (size_t)TROWS * TCOLS;
  unsigned short* xlo = xhi + (size_t)RROWS * TCOLS;
  unsigned*       vws = (unsigned*)(xlo + (size_t)RROWS * TCOLS);

  trig_init<<<dim3((TROWS * TCOLS) / 256), 256, 0, stream>>>(Thi, Tlo);
  split_x<<<dim3((RROWS * (TCOLS / 2)) / 256), 256, 0, stream>>>(x, xhi, xlo);
  dft_gemm<<<dim3(12, 360), 256, 0, stream>>>(xhi, xlo, Thi, Tlo, vws);
  leg_gemm<<<dim3(6, MMAX), 256, 0, stream>>>(vws, wgt, outp);
}
